// Diffuser_47656957116662
// MI455X (gfx1250) — compile-verified, matrix operand order checked by reading, not run on hardware
//
#include <hip/hip_runtime.h>
#include <math.h>

typedef __attribute__((ext_vector_type(16))) _Float16 v16h;
typedef __attribute__((ext_vector_type(8)))  float    v8f;
typedef __attribute__((ext_vector_type(4)))  unsigned int v4u;
typedef __attribute__((ext_vector_type(8)))  int      v8i;
typedef __attribute__((ext_vector_type(4)))  int      v4i;

#define N_NODES   128
#define B_GRAPH   128
#define DIM_IN    64
#define HID_R     128
#define NUM_STACK 17
#define EDGE_DIM  32
#define EPS       1e-5f

// k index for A/B fragment element e (0..15), hi = lane>=16
__device__ __forceinline__ int fkmap(int e, int hi) {
  return (e < 8) ? (hi * 8 + e) : (16 + hi * 8 + (e - 8));
}

__device__ __forceinline__ v8f wmma_f16(v16h a, v16h b, v8f c) {
  return __builtin_amdgcn_wmma_f32_16x16x32_f16(false, a, false, b, (short)0, c,
                                                false, false);
}

// --- CDNA5 async global->LDS copy (ISA 08_async_tensor §4), ASYNCcnt tracked.
// lds_addr = low 32 bits of the generic LDS pointer.
__device__ __forceinline__ void async_g2l_b128(unsigned int lds_addr, const void* gaddr) {
  asm volatile("global_load_async_to_lds_b128 %0, %1, off"
               :: "v"(lds_addr), "v"(gaddr) : "memory");
}
__device__ __forceinline__ void wait_async0() {
  asm volatile("s_wait_asynccnt 0" ::: "memory");
}

// ---------------- K1: generic f32 GEMM via f16 WMMA:  out = A[M,K] @ W[K,Nc] + bias
__global__ void __launch_bounds__(32)
k_gemm_f32(const float* __restrict__ A, const float* __restrict__ W,
           const float* __restrict__ bias, float* __restrict__ out,
           int M, int K, int Nc) {
  int tm = blockIdx.x, tn = blockIdx.y;
  int lane = threadIdx.x & 31, hi = lane >> 4, l15 = lane & 15;
  v8f acc = {};
  int row = tm * 16 + l15;
  int col = tn * 16 + l15;
  for (int kt = 0; kt < K; kt += 32) {
    v16h af, bf;
    const float* ar = A + (size_t)row * K + kt;
#pragma unroll
    for (int e = 0; e < 16; ++e) {
      int k = fkmap(e, hi);
      af[e] = (_Float16)ar[k];
      bf[e] = (_Float16)W[(size_t)(kt + k) * Nc + col];
    }
    acc = wmma_f16(af, bf, acc);
  }
  float bv = bias ? bias[col] : 0.f;
#pragma unroll
  for (int r = 0; r < 8; ++r) {
    int orow = tm * 16 + r + 8 * hi;
    out[(size_t)orow * Nc + col] = acc[r] + bv;
  }
}

// ---------------- K2: e = edge_attr @ C_w + C_b + Ax[dst] + Ax[src]
__global__ void __launch_bounds__(32)
k_edge_e(const float* __restrict__ ea, const float* __restrict__ Cw,
         const float* __restrict__ Cb, const float* __restrict__ Ax,
         const int* __restrict__ ei, float* __restrict__ eo, int E) {
  int tm = blockIdx.x, tn = blockIdx.y;
  int lane = threadIdx.x & 31, hi = lane >> 4, l15 = lane & 15;
  v8f acc = {};
  int row = tm * 16 + l15;
  int col = tn * 16 + l15;
  for (int kt = 0; kt < DIM_IN; kt += 32) {
    v16h af, bf;
    const float* ar = ea + (size_t)row * DIM_IN + kt;
#pragma unroll
    for (int e = 0; e < 16; ++e) {
      int k = fkmap(e, hi);
      af[e] = (_Float16)ar[k];
      bf[e] = (_Float16)Cw[(size_t)(kt + k) * HID_R + col];
    }
    acc = wmma_f16(af, bf, acc);
  }
  float bv = Cb[col];
  const int* srcp = ei;
  const int* dstp = ei + E;
#pragma unroll
  for (int r = 0; r < 8; ++r) {
    int orow = tm * 16 + r + 8 * hi;
    int s = srcp[orow], d = dstp[orow];
    float v = acc[r] + bv + Ax[(size_t)d * HID_R + col] + Ax[(size_t)s * HID_R + col];
    eo[(size_t)orow * HID_R + col] = v;
  }
}

// ---------------- K3: per-channel sum / sumsq over e  (st pre-zeroed)
__global__ void __launch_bounds__(256)
k_e_stats(const float* __restrict__ e, float* __restrict__ st, int E) {
  int c = threadIdx.x & 127;
  int sub = threadIdx.x >> 7;
  float s = 0.f, s2 = 0.f;
  for (int r = blockIdx.x * 2 + sub; r < E; r += gridDim.x * 2) {
    __builtin_prefetch(&e[(size_t)(r + gridDim.x * 2) * HID_R + c], 0, 1);
    float v = e[(size_t)r * HID_R + c];
    s += v; s2 += v * v;
  }
  __shared__ float sh[256];
  sh[threadIdx.x] = s;
  __syncthreads();
  if (sub == 0) atomicAdd(&st[c], sh[c] + sh[c + 128]);
  __syncthreads();
  sh[threadIdx.x] = s2;
  __syncthreads();
  if (sub == 0) atomicAdd(&st[128 + c], sh[c] + sh[c + 128]);
}

__global__ void k_r_final(const float* __restrict__ st, const float* __restrict__ g,
                          const float* __restrict__ b, float* __restrict__ ss, int E) {
  int c = threadIdx.x;
  float mu = st[c] / (float)E;
  float var = st[128 + c] / (float)E - mu * mu;
  float sc = rsqrtf(var + EPS) * g[c];
  ss[c] = sc;
  ss[128 + c] = b[c] - mu * sc;
}

// ---------------- K5: normalize + relu + proj + sigmoid -> w_e  (one wave / edge)
__global__ void __launch_bounds__(256)
k_edge_w(const float* __restrict__ e, const float* __restrict__ ss,
         const float* __restrict__ pw, const float* __restrict__ pb,
         float* __restrict__ we, int E) {
  int wave = blockIdx.x * (blockDim.x / 32) + (threadIdx.x / 32);
  int lane = threadIdx.x & 31;
  if (wave >= E) return;
  float acc = 0.f;
  for (int c = lane; c < HID_R; c += 32) {
    float v = e[(size_t)wave * HID_R + c] * ss[c] + ss[128 + c];
    v = fmaxf(v, 0.f);
    acc += v * pw[c];
  }
  for (int off = 16; off; off >>= 1) acc += __shfl_down(acc, off, 32);
  if (lane == 0) {
    float z = acc + pb[0];
    we[wave] = 1.f / (1.f + __expf(-z));
  }
}

// ---------------- K6: scatter edge weights into dense adjacency (adj pre-zeroed)
__global__ void __launch_bounds__(256)
k_scatter(const int* __restrict__ ei, const float* __restrict__ we,
          float* __restrict__ adj, int E) {
  int t = blockIdx.x * blockDim.x + threadIdx.x;
  if (t >= E) return;
  int s = ei[t], d = ei[E + t];
  int g = s / N_NODES;
  int ls = s - g * N_NODES;
  int ld = d - (d / N_NODES) * N_NODES;
  atomicAdd(&adj[((size_t)g * N_NODES + ls) * N_NODES + ld], we[t]);
}

// ---------------- K7: row-normalize; emit stack ch0 (identity), ch1 (P f16), and P^T f16
__global__ void __launch_bounds__(256)
k_rownorm(const float* __restrict__ adj, _Float16* __restrict__ stk,
          _Float16* __restrict__ Ptg) {
  int row = blockIdx.x * 8 + (threadIdx.x >> 5);
  int lane = threadIdx.x & 31;
  int b = row >> 7, i = row & 127;
  size_t base = (size_t)row * N_NODES;
  float s = 0.f;
  for (int j = lane; j < N_NODES; j += 32) s += adj[base + j];
  for (int off = 16; off; off >>= 1) s += __shfl_down(s, off, 32);
  s = __shfl(s, 0, 32);
  float inv = (s == 0.f) ? 1.f : (1.f / s);
  size_t sb0 = (((size_t)b * NUM_STACK + 0) * N_NODES + i) * N_NODES;
  size_t sb1 = (((size_t)b * NUM_STACK + 1) * N_NODES + i) * N_NODES;
  size_t ptb = (size_t)b * N_NODES * N_NODES;
  for (int j = lane; j < N_NODES; j += 32) {
    float p = adj[base + j] * inv;
    _Float16 h = (_Float16)p;
    stk[sb1 + j] = h;
    stk[sb0 + j] = (_Float16)((i == j) ? 1.f : 0.f);
    Ptg[ptb + (size_t)j * N_NODES + i] = h;
  }
}

// ---------------- K8: matrix powers P^2..P^16; async global->LDS staging, WMMA chain
__global__ void __launch_bounds__(256)
k_powers(const _Float16* __restrict__ Ptg, _Float16* __restrict__ stk) {
  __shared__ alignas(16) _Float16 Pt[N_NODES * N_NODES];  // P^T: Pt[n*128+k]=P[k][n]
  __shared__ alignas(16) _Float16 Pk[N_NODES * N_NODES];  // current power, row-major
  int b = blockIdx.x;
  int tid = threadIdx.x;

#if __has_builtin(__builtin_amdgcn_tensor_load_to_lds) && __has_builtin(__builtin_amdgcn_s_wait_tensorcnt)
  // NULL-descriptor TDM op (D# group0.count==0 -> NULL tensor, architectural no-op)
  if (tid < 32) {
    v4u g0 = {}; v8i g1 = {}; v4i g2 = {}, g3 = {};
#if __has_include(<hip/amd_detail/amd_gfx1250_TDM.h>)
    v8i g4 = {};
    __builtin_amdgcn_tensor_load_to_lds(g0, g1, g2, g3, g4, 0);
#else
    __builtin_amdgcn_tensor_load_to_lds(g0, g1, g2, g3, 0);
#endif
    __builtin_amdgcn_s_wait_tensorcnt(0);
  }
#endif

  // stage P (stack channel 1, f16 row-major) and P^T into LDS with async B128 copies
  const char* Pg1 = (const char*)(stk + ((size_t)b * NUM_STACK + 1) * N_NODES * N_NODES);
  const char* Ptgb = (const char*)(Ptg + (size_t)b * N_NODES * N_NODES);
  unsigned int pk_base = (unsigned int)(size_t)&Pk[0];
  unsigned int pt_base = (unsigned int)(size_t)&Pt[0];
  for (int c = tid; c < 2048; c += 256) {           // 2048 x 16B = 32KB per buffer
    async_g2l_b128(pk_base + c * 16, Pg1 + (size_t)c * 16);
    async_g2l_b128(pt_base + c * 16, Ptgb + (size_t)c * 16);
  }
  wait_async0();
  __syncthreads();

#if __has_builtin(__builtin_amdgcn_cluster_id_x) && __has_builtin(__builtin_amdgcn_s_cluster_barrier)
  if (__builtin_amdgcn_cluster_id_x() != 0) __builtin_amdgcn_s_cluster_barrier();
#endif

  int wave = tid >> 5, lane = tid & 31, hi = lane >> 4, l15 = lane & 15;
  int row = wave * 16 + l15;
  for (int step = 2; step <= 16; ++step) {
    // preload the wave's 4 A fragments (its 16-row block of Pk, K=128)
    v16h afr[4];
#pragma unroll
    for (int kt = 0; kt < 4; ++kt) {
      const _Float16* ar = &Pk[row * N_NODES + kt * 32];
#pragma unroll
      for (int e = 0; e < 16; ++e) afr[kt][e] = ar[fkmap(e, hi)];
    }
    v8f accs[8];
#pragma unroll 1
    for (int tn = 0; tn < 8; ++tn) {
      v8f acc = {};
      int col = tn * 16 + l15;
#pragma unroll
      for (int kt = 0; kt < 4; ++kt) {
        v16h bf;
        const _Float16* br = &Pt[col * N_NODES + kt * 32];
#pragma unroll
        for (int e = 0; e < 16; ++e) bf[e] = br[fkmap(e, hi)];
        acc = wmma_f16(afr[kt], bf, acc);
      }
      accs[tn] = acc;
    }
    __syncthreads();  // everyone done reading Pk
    size_t sbase = ((size_t)b * NUM_STACK + step) * N_NODES * N_NODES;
#pragma unroll 1
    for (int tn = 0; tn < 8; ++tn) {
      int col = tn * 16 + l15;
#pragma unroll
      for (int r = 0; r < 8; ++r) {
        int orow = wave * 16 + r + 8 * hi;
        _Float16 h = (_Float16)accs[tn][r];
        Pk[orow * N_NODES + col] = h;
        stk[sbase + (size_t)orow * N_NODES + col] = h;
      }
    }
    __syncthreads();
  }
}

// ---------------- K9: stack BN stats per channel (st pre-zeroed)
__global__ void __launch_bounds__(256)
k_s_stats(const _Float16* __restrict__ stk, float* __restrict__ st) {
  int ch = blockIdx.y;
  int tid = threadIdx.x;
  const int per = N_NODES * N_NODES;
  float s = 0.f, s2 = 0.f;
  for (int i = blockIdx.x * 256 + tid; i < B_GRAPH * per; i += gridDim.x * 256) {
    int b = i / per;
    int pos = i - b * per;
    float v = (float)stk[((size_t)b * NUM_STACK + ch) * per + pos];
    s += v; s2 += v * v;
  }
  __shared__ float sh[256];
  sh[tid] = s;
  __syncthreads();
  for (int o = 128; o > 0; o >>= 1) { if (tid < o) sh[tid] += sh[tid + o]; __syncthreads(); }
  if (tid == 0) atomicAdd(&st[ch], sh[0]);
  __syncthreads();
  sh[tid] = s2;
  __syncthreads();
  for (int o = 128; o > 0; o >>= 1) { if (tid < o) sh[tid] += sh[tid + o]; __syncthreads(); }
  if (tid == 0) atomicAdd(&st[NUM_STACK + ch], sh[0]);
}

__global__ void k_s_final(const float* __restrict__ st, const float* __restrict__ g,
                          const float* __restrict__ b, float* __restrict__ ss) {
  int c = threadIdx.x;
  float cnt = (float)B_GRAPH * (float)N_NODES * (float)N_NODES;
  float mu = st[c] / cnt;
  float var = st[NUM_STACK + c] / cnt - mu * mu;
  float sc = rsqrtf(var + EPS) * g[c];
  ss[c] = sc;
  ss[NUM_STACK + c] = b[c] - mu * sc;
}

// ---------------- K11: fused final MLP  [2M x 17(pad32)] -> relu 32 -> 32, NT stores
__global__ void __launch_bounds__(256)
k_mlp(const _Float16* __restrict__ stk, const float* __restrict__ ss,
      const float* __restrict__ l1w, const float* __restrict__ l1b,
      const float* __restrict__ l2w, const float* __restrict__ l2b,
      float* __restrict__ out) {
  __shared__ alignas(16) _Float16 hbuf[8 * 16 * 32];
  int wave = threadIdx.x >> 5;
  int tm = blockIdx.x * 8 + wave;               // 16-row tile of the 2M-row matrix
  int lane = threadIdx.x & 31, hi = lane >> 4, l15 = lane & 15;
  int R = tm * 16 + l15;
  int bI = R >> 14;                             // / (N*N)
  int rem = R & 16383;                          // i*128 + j
  // stage 1: A = normalized stack (K padded 17 -> 32), B = l1_w (padded)
  v16h af, bf0, bf1;
#pragma unroll
  for (int e = 0; e < 16; ++e) {
    int k = fkmap(e, hi);
    float av = 0.f; _Float16 b0 = (_Float16)0.f, b1 = (_Float16)0.f;
    if (k < NUM_STACK) {
      float raw = (float)stk[(((size_t)bI * NUM_STACK + k) << 14) + rem];
      av = raw * ss[k] + ss[NUM_STACK + k];
      b0 = (_Float16)l1w[k * EDGE_DIM + l15];
      b1 = (_Float16)l1w[k * EDGE_DIM + 16 + l15];
    }
    af[e] = (_Float16)av; bf0[e] = b0; bf1[e] = b1;
  }
  v8f z = {};
  v8f h0 = wmma_f16(af, bf0, z);
  v8f h1 = wmma_f16(af, bf1, z);
  float lb0 = l1b[l15], lb1 = l1b[16 + l15];
  _Float16* hb = &hbuf[wave * 512];
#pragma unroll
  for (int r = 0; r < 8; ++r) {
    int rr = r + 8 * hi;
    hb[rr * 32 + l15]      = (_Float16)fmaxf(h0[r] + lb0, 0.f);
    hb[rr * 32 + 16 + l15] = (_Float16)fmaxf(h1[r] + lb1, 0.f);
  }
  __syncthreads();
  // stage 2: A = h (16x32 from LDS), B = l2_w
  v16h a2, c0, c1;
  const _Float16* hr = &hb[l15 * 32];
#pragma unroll
  for (int e = 0; e < 16; ++e) {
    int k = fkmap(e, hi);
    a2[e] = hr[k];
    c0[e] = (_Float16)l2w[k * EDGE_DIM + l15];
    c1[e] = (_Float16)l2w[k * EDGE_DIM + 16 + l15];
  }
  v8f o0 = wmma_f16(a2, c0, z);
  v8f o1 = wmma_f16(a2, c1, z);
  float ob0 = l2b[l15], ob1 = l2b[16 + l15];
#pragma unroll
  for (int r = 0; r < 8; ++r) {
    int orow = tm * 16 + r + 8 * hi;
    __builtin_nontemporal_store(o0[r] + ob0, &out[(size_t)orow * EDGE_DIM + l15]);
    __builtin_nontemporal_store(o1[r] + ob1, &out[(size_t)orow * EDGE_DIM + 16 + l15]);
  }
}

extern "C" void kernel_launch(void* const* d_in, const int* in_sizes, int n_in,
                              void* d_out, int out_size, void* d_ws, size_t ws_size,
                              hipStream_t stream) {
  const float* x   = (const float*)d_in[0];
  const float* ea  = (const float*)d_in[1];
  const int*   ei  = (const int*)d_in[2];
  // d_in[3] = batch (implied by contiguous layout, unused)
  const float* Aw  = (const float*)d_in[4];
  const float* Ab  = (const float*)d_in[5];
  const float* Cw  = (const float*)d_in[6];
  const float* Cb  = (const float*)d_in[7];
  const float* pw  = (const float*)d_in[8];
  const float* pb  = (const float*)d_in[9];
  const float* brg = (const float*)d_in[10];
  const float* brb = (const float*)d_in[11];
  const float* bsg = (const float*)d_in[12];
  const float* bsb = (const float*)d_in[13];
  const float* l1w = (const float*)d_in[14];
  const float* l1b = (const float*)d_in[15];
  const float* l2w = (const float*)d_in[16];
  const float* l2b = (const float*)d_in[17];

  const int E  = in_sizes[2] / 2;            // 131072
  const int Mn = in_sizes[0] / DIM_IN;       // 16384 nodes total

  char* ws = (char*)d_ws;
  size_t o = 0;
  float*    Ax   = (float*)(ws + o);  o += (size_t)Mn * HID_R * 4;
  float*    eB   = (float*)(ws + o);  o += (size_t)E * HID_R * 4;
  float*    we   = (float*)(ws + o);  o += (size_t)E * 4;
  float*    adj  = (float*)(ws + o);  o += (size_t)B_GRAPH * N_NODES * N_NODES * 4;
  _Float16* stk  = (_Float16*)(ws + o);
  o += (size_t)B_GRAPH * NUM_STACK * N_NODES * N_NODES * 2;
  _Float16* Ptg  = (_Float16*)(ws + o);
  o += (size_t)B_GRAPH * N_NODES * N_NODES * 2;
  float*    st_r = (float*)(ws + o);  o += 256 * 4;
  float*    ss_r = (float*)(ws + o);  o += 256 * 4;
  float*    st_s = (float*)(ws + o);  o += 64 * 4;
  float*    ss_s = (float*)(ws + o);  o += 64 * 4;

  hipMemsetAsync(st_r, 0, 256 * 4, stream);
  hipMemsetAsync(st_s, 0, 64 * 4, stream);
  hipMemsetAsync(adj, 0, (size_t)B_GRAPH * N_NODES * N_NODES * 4, stream);

  k_gemm_f32<<<dim3(Mn / 16, HID_R / 16), 32, 0, stream>>>(x, Aw, Ab, Ax, Mn, DIM_IN, HID_R);
  k_edge_e<<<dim3(E / 16, HID_R / 16), 32, 0, stream>>>(ea, Cw, Cb, Ax, ei, eB, E);
  k_e_stats<<<1024, 256, 0, stream>>>(eB, st_r, E);
  k_r_final<<<1, 128, 0, stream>>>(st_r, brg, brb, ss_r, E);
  k_edge_w<<<E / 8, 256, 0, stream>>>(eB, ss_r, pw, pb, we, E);
  k_scatter<<<(E + 255) / 256, 256, 0, stream>>>(ei, we, adj, E);
  k_rownorm<<<(B_GRAPH * N_NODES) / 8, 256, 0, stream>>>(adj, stk, Ptg);
  k_powers<<<B_GRAPH, 256, 0, stream>>>(Ptg, stk);
  k_s_stats<<<dim3(64, NUM_STACK), 256, 0, stream>>>(stk, st_s);
  k_s_final<<<1, NUM_STACK, 0, stream>>>(st_s, bsg, bsb, ss_s);
  k_mlp<<<(B_GRAPH * N_NODES * N_NODES / 16) / 8, 256, 0, stream>>>(
      stk, ss_s, l1w, l1b, l2w, l2b, (float*)d_out);
  (void)n_in; (void)out_size; (void)ws_size;
}
